// EfficientFeatureMatching_12068858101979
// MI455X (gfx1250) — matrix operand order repaired and checked by reading, not compiled, not run on hardware
//
#include <hip/hip_runtime.h>

// ---------------------------------------------------------------------------
// EfficientFeatureMatching forward for gfx1250 (MI455X).
// Conv layers = implicit GEMM via v_wmma_f32_16x16x32_f16 (f16 in, f32 acc).
// Activations NHWC f16, weights pre-transformed to [Cout][tap][Cin] f16 so
// both WMMA fragments are contiguous b128 loads (CDNA5 ISA 7.12.2 striping).
// Wave tile = 32 outch x 64 pixels (8 independent accumulators -> no WMMA
// RAW hazards, 1.5 b128 loads per WMMA). K-loop fully unrolled (Cin is a
// template parameter); tile decode is pure blockIdx (no runtime division).
// ---------------------------------------------------------------------------

typedef __attribute__((ext_vector_type(16))) _Float16 v16h;
typedef __attribute__((ext_vector_type(8)))  float    v8f;
typedef _Float16 f16_t;

#define H_   144
#define W_   144
#define PIX_ (H_ * W_)      // 20736; 16 | W_ so 16-pixel tiles never cross rows
#define PT   4              // 16-pixel tiles per wave
#define MT   2              // 16-outch tiles per wave
// pixel groups = PIX_/(16*PT) = 324 = 81 blocks * 4 waves

union FragAB { v16h v; uint4 q[2]; };
union Pack8  { f16_t h[8]; uint4 q; };

// ---------------------------------------------------------------------------
// WMMA implicit-GEMM conv. TAPS = 9 (3x3 pad1) or 1 (1x1); CIN compile-time.
//   X  : [N][H][W][CIN]  f16 (NHWC)
//   Wt : [Cout][TAPS][CIN] f16 (pre-transformed)
//   Y  : [N][H][W][Cstride] f16, written at channel offset chOff
//   post-op: relu((acc + bias)*scale + shift); scale/shift null => 1/0
// Grid: (81, coutTiles/MT, N), block 128 (4 waves). Wave = 32 outch x 64 pix.
// ---------------------------------------------------------------------------
template <int TAPS, int CIN>
__global__ __launch_bounds__(128) void conv_wmma_kernel(
    const f16_t* __restrict__ X, const f16_t* __restrict__ Wt,
    const float* __restrict__ bias, const float* __restrict__ scale,
    const float* __restrict__ shift, f16_t* __restrict__ Y,
    int Cstride, int chOff)
{
  constexpr int CHUNKS = CIN / 32;               // K-chunks of 32 within one tap
  const int lane = threadIdx.x & 31;
  const int hlf  = lane >> 4;   // lane half selects K sub-range per ISA layout
  const int l15  = lane & 15;
  const int pg    = blockIdx.x * 4 + (threadIdx.x >> 5);   // 0..323
  const int mBase = blockIdx.y * (16 * MT);
  const int n     = blockIdx.z;

  // per-tile pixel coordinates for this lane (B-matrix column = pixel)
  int ph[PT], pw[PT];
#pragma unroll
  for (int t = 0; t < PT; ++t) {
    int p = pg * (16 * PT) + t * 16 + l15;
    ph[t] = p / W_;                              // constant divisor -> mul/shift
    pw[t] = p - ph[t] * W_;
  }

  v8f zero = {};
  v8f acc[MT][PT];
#pragma unroll
  for (int j = 0; j < MT; ++j)
#pragma unroll
    for (int t = 0; t < PT; ++t) acc[j][t] = zero;

  const f16_t* Xn = X + (size_t)n * PIX_ * CIN;
  // A fragment base: row m = mBase+16j+l15; lane-half picks K runs {8h.., 16+8h..}
  const f16_t* arowBase[MT];
#pragma unroll
  for (int j = 0; j < MT; ++j)
    arowBase[j] = Wt + ((size_t)(mBase + 16 * j + l15) * TAPS) * CIN + 8 * hlf;

  for (int tap = 0; tap < TAPS; ++tap) {
    const int ky = (TAPS == 9) ? (tap / 3 - 1) : 0;
    const int kx = (TAPS == 9) ? (tap - (tap / 3) * 3 - 1) : 0;
    if (TAPS == 9 && tap + 1 < TAPS) {           // warm L0 for next tap's A panels
      __builtin_prefetch(arowBase[0] + (tap + 1) * CIN, 0, 3);
      __builtin_prefetch(arowBase[1] + (tap + 1) * CIN, 0, 3);
    }

    const f16_t* brow[PT];
    bool bval[PT];
#pragma unroll
    for (int t = 0; t < PT; ++t) {
      int ih = ph[t] + ky, iw = pw[t] + kx;
      bval[t] = ((unsigned)ih < H_) & ((unsigned)iw < W_);
      // B fragment: column = pixel, K runs {8h.., 16+8h..} matching A
      brow[t] = Xn + ((size_t)ih * W_ + iw) * CIN + 8 * hlf;
    }

#pragma unroll
    for (int cb = 0; cb < CHUNKS; ++cb) {
      FragAB a[MT];
#pragma unroll
      for (int j = 0; j < MT; ++j) {
        const f16_t* arow = arowBase[j] + tap * CIN + cb * 32;
        a[j].q[0] = *(const uint4*)(arow);       // K = 8h .. 8h+7
        a[j].q[1] = *(const uint4*)(arow + 16);  // K = 16+8h .. 16+8h+7
      }
#pragma unroll
      for (int t = 0; t < PT; ++t) {
        FragAB b;
        if (bval[t]) {
          b.q[0] = *(const uint4*)(brow[t] + cb * 32);      // K = 8h .. 8h+7
          b.q[1] = *(const uint4*)(brow[t] + cb * 32 + 16); // K = 16+8h ..
        } else {
          b.q[0] = make_uint4(0, 0, 0, 0);
          b.q[1] = make_uint4(0, 0, 0, 0);
        }
#pragma unroll
        for (int j = 0; j < MT; ++j)
          acc[j][t] = __builtin_amdgcn_wmma_f32_16x16x32_f16(
              false, a[j].v, false, b.v, (short)0, acc[j][t], false, false);
      }
    }
  }

  // Epilogue: C/D layout => lane holds rows m = 8*hlf + r, col = l15.
  // 8 consecutive channels -> one packed b128 store per (j, t).
#pragma unroll
  for (int j = 0; j < MT; ++j) {
    const int chL = mBase + 16 * j + 8 * hlf;
    float bb[8], ss[8], tt[8];
#pragma unroll
    for (int r = 0; r < 8; ++r) {
      bb[r] = bias[chL + r];
      ss[r] = scale ? scale[chL + r] : 1.f;
      tt[r] = shift ? shift[chL + r] : 0.f;
    }
#pragma unroll
    for (int t = 0; t < PT; ++t) {
      int p = pg * (16 * PT) + t * 16 + l15;
      Pack8 st;
#pragma unroll
      for (int r = 0; r < 8; ++r) {
        float v = (acc[j][t][r] + bb[r]) * ss[r] + tt[r];
        st.h[r] = (f16_t)(v > 0.f ? v : 0.f);
      }
      *(uint4*)(Y + ((size_t)n * PIX_ + p) * Cstride + chOff + chL) = st.q;
    }
  }
}

// OIHW f32 -> [Cout][tap][CinP] f16 (zero pad cin >= Cin, e.g. lc1 81->96)
template <int CinI, int CinP, int T>
__global__ void xform_w_kernel(const float* __restrict__ w, f16_t* __restrict__ wT,
                               int tot)
{
  int idx = blockIdx.x * blockDim.x + threadIdx.x;
  if (idx >= tot) return;
  int c   = idx % CinP;                          // constant divisors
  int tap = (idx / CinP) % T;
  int m   = idx / (CinP * T);
  wT[idx] = (c < CinI) ? (f16_t)w[((size_t)m * CinI + c) * T + tap] : (f16_t)0.f;
}

// NCHW f32 -> NHWC f16 with nearest-neighbor upsample factor R
template <int C, int Hin, int Win, int R>
__global__ void to_nhwc_kernel(const float* __restrict__ in, f16_t* __restrict__ out,
                               int total)
{
  int idx = blockIdx.x * blockDim.x + threadIdx.x;
  if (idx >= total) return;
  int c = idx % C;
  int rest = idx / C;
  constexpr int Wo = Win * R, Ho = Hin * R;
  int w = rest % Wo; rest /= Wo;
  int h = rest % Ho;
  int n = rest / Ho;
  out[idx] = (f16_t)in[(((size_t)n * C + c) * Hin + h / R) * Win + (w / R)];
}

// cost[k,p] = sum_c f1[p,c] * f2[p+shift(k),c], 9x9 window pad 4.
// COST stored NHWC with 96 channels (81 real + 15 zero pad for WMMA K).
__global__ void cost_volume_kernel(const f16_t* __restrict__ FU, f16_t* __restrict__ COST)
{
  int idx = blockIdx.x * blockDim.x + threadIdx.x;
  if (idx >= PIX_ * 96) return;
  int k = idx % 96;
  int p = idx / 96;
  if (k >= 81) { COST[idx] = (f16_t)0.f; return; }
  int h = p / W_;
  int w = p - h * W_;
  int ih = h + k / 9 - 4;
  int iw = w + (k - (k / 9) * 9) - 4;
  float acc = 0.f;
  if ((unsigned)ih < H_ && (unsigned)iw < W_) {
    const uint4* a4 = (const uint4*)(FU + (size_t)p * 192);
    const uint4* b4 = (const uint4*)(FU + (size_t)PIX_ * 192 + ((size_t)ih * W_ + iw) * 192);
#pragma unroll 4
    for (int q = 0; q < 24; ++q) {               // 24 x b128 = 192 f16 per side
      Pack8 A, B;
      A.q = a4[q]; B.q = b4[q];
#pragma unroll
      for (int r = 0; r < 8; ++r) acc += (float)A.h[r] * (float)B.h[r];
    }
  }
  COST[idx] = (f16_t)acc;
}

// copy f1 (FU batch 0, 192 ch) into AI channels 0..191 (256-ch NHWC), b128 copies
__global__ void copy_f1_kernel(const f16_t* __restrict__ FU, f16_t* __restrict__ AI)
{
  int idx = blockIdx.x * blockDim.x + threadIdx.x;
  if (idx >= PIX_ * 24) return;            // 24 uint4 (=192 f16) per pixel
  int q = idx % 24;
  int p = idx / 24;
  ((uint4*)(AI + (size_t)p * 256))[q] = ((const uint4*)(FU + (size_t)p * 192))[q];
}

// att[p] = sigmoid( sum_c AT1[p,c]*w2[c] + b2 )
__global__ void att_kernel(const f16_t* __restrict__ AT1, const float* __restrict__ w2,
                           const float* __restrict__ b2, float* __restrict__ ATT)
{
  int p = blockIdx.x * blockDim.x + threadIdx.x;
  if (p >= PIX_) return;
  float acc = b2[0];
  const uint4* x4 = (const uint4*)(AT1 + (size_t)p * 128);
#pragma unroll 4
  for (int q = 0; q < 16; ++q) {                 // 16 x b128 = 128 f16
    Pack8 A;
    A.q = x4[q];
#pragma unroll
    for (int r = 0; r < 8; ++r) acc += (float)A.h[r] * w2[q * 8 + r];
  }
  ATT[p] = 1.f / (1.f + __expf(-acc));
}

// out (NCHW f32, 1x256x144x144): ch<192 -> |f1-f2|*att ; ch>=192 -> corr*att
__global__ void final_kernel(const f16_t* __restrict__ FU, const f16_t* __restrict__ AI,
                             const float* __restrict__ ATT, float* __restrict__ out)
{
  int idx = blockIdx.x * blockDim.x + threadIdx.x;
  if (idx >= 256 * PIX_) return;
  int p  = idx % PIX_;
  int ch = idx / PIX_;
  float a = ATT[p];
  float v;
  if (ch < 192) {
    float f1 = (float)FU[(size_t)p * 192 + ch];
    float f2 = (float)FU[(size_t)PIX_ * 192 + (size_t)p * 192 + ch];
    v = fabsf(f1 - f2);
  } else {
    v = (float)AI[(size_t)p * 256 + 192 + (ch - 192)];   // corr lives in AI[192..255]
  }
  out[idx] = v * a;
}

extern "C" void kernel_launch(void* const* d_in, const int* in_sizes, int n_in,
                              void* d_out, int out_size, void* d_ws, size_t ws_size,
                              hipStream_t stream)
{
  const float* features = (const float*)d_in[0];
  const float* dino     = (const float*)d_in[1];
  const float* up_w = (const float*)d_in[2];  const float* up_b = (const float*)d_in[3];
  const float* up_s = (const float*)d_in[4];  const float* up_t = (const float*)d_in[5];
  const float* cp_w = (const float*)d_in[6];  const float* cp_b = (const float*)d_in[7];
  const float* cp_s = (const float*)d_in[8];  const float* cp_t = (const float*)d_in[9];
  const float* dp_w = (const float*)d_in[10]; const float* dp_b = (const float*)d_in[11];
  const float* dp_s = (const float*)d_in[12]; const float* dp_t = (const float*)d_in[13];
  const float* fu_w = (const float*)d_in[14]; const float* fu_b = (const float*)d_in[15];
  const float* fu_s = (const float*)d_in[16]; const float* fu_t = (const float*)d_in[17];
  const float* lc1_w = (const float*)d_in[18]; const float* lc1_b = (const float*)d_in[19];
  const float* lc2_w = (const float*)d_in[20]; const float* lc2_b = (const float*)d_in[21];
  const float* at1_w = (const float*)d_in[22]; const float* at1_b = (const float*)d_in[23];
  const float* at2_w = (const float*)d_in[24]; const float* at2_b = (const float*)d_in[25];

  const size_t P = PIX_;
  char* cur = (char*)d_ws;
  auto carve = [&](size_t bytes) -> char* {
    char* r = cur; cur += (bytes + 255) & ~(size_t)255; return r;
  };
  f16_t* X0 = (f16_t*)carve(2 * P * 128 * 2);   // upsampled features (NHWC f16)
  f16_t* X1 = (f16_t*)carve(2 * P * 128 * 2);   // after up conv
  f16_t* Dn = (f16_t*)carve(2 * P * 64 * 2);    // dino NHWC f16
  f16_t* Fb = (f16_t*)carve(2 * P * 192 * 2);   // concat(cp, dp)
  f16_t* FU = (f16_t*)carve(2 * P * 192 * 2);   // fused
  f16_t* wTup  = (f16_t*)carve((size_t)128 * 9 * 128 * 2);
  f16_t* wTcp  = (f16_t*)carve((size_t)96 * 128 * 2);
  f16_t* wTdp  = (f16_t*)carve((size_t)96 * 64 * 2);
  f16_t* wTfu  = (f16_t*)carve((size_t)192 * 9 * 192 * 2);
  f16_t* wTlc1 = (f16_t*)carve((size_t)64 * 96 * 2);
  f16_t* wTlc2 = (f16_t*)carve((size_t)64 * 9 * 64 * 2);
  f16_t* wTat1 = (f16_t*)carve((size_t)128 * 9 * 256 * 2);
  // Lifetime-disjoint aliases (producers of the alias run after the last
  // reader of the original buffer on the same stream):
  f16_t* AI    = X0;                                        // 256-ch NHWC, after up
  f16_t* AT1   = X1;                                        // 128-ch, after cp
  f16_t* COST  = Fb;                                        // 96-ch, after fu
  f16_t* CORR1 = (f16_t*)((char*)Fb + P * 96 * 2);          // 64-ch
  float* ATT   = (float*)((char*)Fb + P * 96 * 2 + P * 64 * 2);

  auto cdiv = [](size_t a, size_t b) { return (unsigned)((a + b - 1) / b); };

  // --- weight transforms (independent) ---
  xform_w_kernel<128,128,9><<<cdiv((size_t)128*9*128,256),256,0,stream>>>(up_w,  wTup,  128*9*128);
  xform_w_kernel<128,128,1><<<cdiv((size_t)96*128,256),    256,0,stream>>>(cp_w,  wTcp,  96*128);
  xform_w_kernel< 64, 64,1><<<cdiv((size_t)96*64,256),     256,0,stream>>>(dp_w,  wTdp,  96*64);
  xform_w_kernel<192,192,9><<<cdiv((size_t)192*9*192,256), 256,0,stream>>>(fu_w,  wTfu,  192*9*192);
  xform_w_kernel< 81, 96,1><<<cdiv((size_t)64*96,256),     256,0,stream>>>(lc1_w, wTlc1, 64*96);
  xform_w_kernel< 64, 64,9><<<cdiv((size_t)64*9*64,256),   256,0,stream>>>(lc2_w, wTlc2, 64*9*64);
  xform_w_kernel<256,256,9><<<cdiv((size_t)128*9*256,256), 256,0,stream>>>(at1_w, wTat1, 128*9*256);

  // --- layout conversions ---
  to_nhwc_kernel<128,36,36,4> <<<cdiv(2*P*128,256),256,0,stream>>>(features, X0, (int)(2*P*128));
  to_nhwc_kernel<64,144,144,1><<<cdiv(2*P*64,256), 256,0,stream>>>(dino,     Dn, (int)(2*P*64));

  // --- WMMA conv pipeline: grid (81 pixel-groups, coutTiles/2, N), 4 waves/blk ---
  conv_wmma_kernel<9,128><<<dim3(81,4,2),128,0,stream>>>(X0, wTup, up_b, up_s, up_t, X1, 128,   0);
  conv_wmma_kernel<1,128><<<dim3(81,3,2),128,0,stream>>>(X1, wTcp, cp_b, cp_s, cp_t, Fb, 192,   0);
  conv_wmma_kernel<1, 64><<<dim3(81,3,2),128,0,stream>>>(Dn, wTdp, dp_b, dp_s, dp_t, Fb, 192,  96);
  conv_wmma_kernel<9,192><<<dim3(81,6,2),128,0,stream>>>(Fb, wTfu, fu_b, fu_s, fu_t, FU, 192,   0);

  cost_volume_kernel<<<cdiv(P*96,256),256,0,stream>>>(FU, COST);
  copy_f1_kernel    <<<cdiv(P*24,256),256,0,stream>>>(FU, AI);

  conv_wmma_kernel<1, 96><<<dim3(81,2,1),128,0,stream>>>(COST,  wTlc1, lc1_b, (const float*)nullptr, (const float*)nullptr, CORR1, 64,   0);
  conv_wmma_kernel<9, 64><<<dim3(81,2,1),128,0,stream>>>(CORR1, wTlc2, lc2_b, (const float*)nullptr, (const float*)nullptr, AI,   256, 192);
  conv_wmma_kernel<9,256><<<dim3(81,4,1),128,0,stream>>>(AI,    wTat1, at1_b, (const float*)nullptr, (const float*)nullptr, AT1,  128,   0);

  att_kernel  <<<cdiv(P,256),    256,0,stream>>>(AT1, at2_w, at2_b, ATT);
  final_kernel<<<cdiv(256*P,256),256,0,stream>>>(FU, AI, ATT, (float*)d_out);
}
